// GATv3Convolution_36627481100820
// MI455X (gfx1250) — matrix operand order repaired and checked by reading, not compiled, not run on hardware
//
#include <hip/hip_runtime.h>
#include <hip/hip_bf16.h>

// ---------------------------------------------------------------------------
// GATv2 x2 + linear skips on gfx1250 (wave32, WMMA f32_16x16x32_f16)
// Edge passes vectorized to b128 loads per lane.
// ---------------------------------------------------------------------------

typedef _Float16 v16h __attribute__((ext_vector_type(16)));
typedef _Float16 v8h  __attribute__((ext_vector_type(8)));
typedef _Float16 v4h  __attribute__((ext_vector_type(4)));
typedef float    v8f  __attribute__((ext_vector_type(8)));
typedef float    v4f  __attribute__((ext_vector_type(4)));

#define FDIM 128
#define NEG_SLOPE 0.2f

__device__ __forceinline__ void atomicMaxF(float* addr, float val) {
  // Monotonic int-ordering trick; addr must be initialized to -inf.
  if (val >= 0.0f) {
    atomicMax((int*)addr, __float_as_int(val));
  } else {
    atomicMin((unsigned int*)addr, __float_as_uint(val));
  }
}

// ---------------- fill / convert / pack -----------------------------------

__global__ void fill_f32(float* __restrict__ p, float v, int n) {
  int t = blockIdx.x * blockDim.x + threadIdx.x;
  if (t < n) p[t] = v;
}

// 4 floats -> 4 halves per thread (n4 = n/4)
__global__ void cvt_f16(const float* __restrict__ x, _Float16* __restrict__ xh, int n4) {
  int t = blockIdx.x * blockDim.x + threadIdx.x;
  if (t >= n4) return;
  v4f xv = ((const v4f*)x)[t];
  v4h o;
#pragma unroll
  for (int i = 0; i < 4; ++i) o[i] = (_Float16)xv[i];
  ((v4h*)xh)[t] = o;
}

// Pack three 128x128 weight matrices (row-major [K,N]) into WMMA-B fragment
// order: elem ((((m*8+ct)*4+kt)*32+lane)*16+i) = W[kt*32+(lane>=16?8:0)+(i<8?i:i+8)][ct*16+(lane&15)]
__global__ void pack_b(const float* __restrict__ W0, const float* __restrict__ W1,
                       const float* __restrict__ W2, _Float16* __restrict__ Bp) {
  int t = blockIdx.x * blockDim.x + threadIdx.x;   // 0 .. 3*8*4*32-1
  if (t >= 3 * 8 * 4 * 32) return;
  int lane = t & 31;
  int kt = (t >> 5) & 3;
  int ct = (t >> 7) & 7;
  int m  = t >> 10;
  const float* W = (m == 0) ? W0 : (m == 1) ? W1 : W2;
  int n  = ct * 16 + (lane & 15);
  int kb = kt * 32 + ((lane >> 4) ? 8 : 0);
  _Float16* out = Bp + (size_t)t * 16;
#pragma unroll
  for (int i = 0; i < 16; ++i)
    out[i] = (_Float16)W[(size_t)(kb + (i < 8 ? i : i + 8)) * FDIM + n];
}

// ---------------- layer-1 GEMMs via WMMA ----------------------------------
// One wave per 16-row strip; reuses A fragments across 3 weights x 8 col tiles.
__global__ __launch_bounds__(256) void gemm3_wmma(
    const _Float16* __restrict__ xh, const _Float16* __restrict__ Bp,
    float* __restrict__ y0, float* __restrict__ y1, float* __restrict__ y2,
    int nstrips) {
  const int lane  = threadIdx.x & 31;
  const int wave  = threadIdx.x >> 5;
  const int strip = blockIdx.x * 8 + wave;
  if (strip >= nstrips) return;
  const int hi   = lane >> 4;                 // 0 or 1
  const int mrow = strip * 16 + (lane & 15);

  // A fragments: 16-bit A 16x32 layout (lanes 0-15: K{0..7,16..23}; 16-31: +8)
  v16h a[4];
  const _Float16* abase = xh + (size_t)mrow * FDIM + hi * 8;
#pragma unroll
  for (int kt = 0; kt < 4; ++kt) {
    v8h lo8 = *(const v8h*)(abase + kt * 32);
    v8h hi8 = *(const v8h*)(abase + kt * 32 + 16);
    a[kt] = __builtin_shufflevector(lo8, hi8, 0,1,2,3,4,5,6,7,8,9,10,11,12,13,14,15);
  }

  const int rowbase = strip * 16 + hi * 8;
#pragma unroll
  for (int m = 0; m < 3; ++m) {
    float* __restrict__ y = (m == 0) ? y0 : (m == 1) ? y1 : y2;
    for (int ct = 0; ct < 8; ++ct) {
      const _Float16* bbase = Bp + ((((size_t)m * 8 + ct) * 4) * 32 + lane) * 16;
      v8f c = {};
#pragma unroll
      for (int kt = 0; kt < 4; ++kt) {
        v16h b = *(const v16h*)(bbase + (size_t)kt * 32 * 16);
        c = __builtin_amdgcn_wmma_f32_16x16x32_f16(
            /*neg_a=*/false, a[kt], /*neg_b=*/false, b,
            /*c_mod=*/(short)0, c, /*reuse_a=*/false, /*reuse_b=*/false);
      }
      const int col = ct * 16 + (lane & 15);
#pragma unroll
      for (int r = 0; r < 8; ++r)
        y[(size_t)(rowbase + r) * FDIM + col] = c[r];
    }
  }
}

// ---------------- edge helpers --------------------------------------------

__device__ __forceinline__ void edge_sd(int e, int E, const int* __restrict__ ei,
                                        int& s, int& d) {
  if (e < E) { s = ei[e]; d = ei[E + e]; }
  else       { s = e - E; d = s; }
}

// layer-1 edge score: wave per edge; lane owns 4 contiguous k (one b128 load each)
__global__ __launch_bounds__(256) void edge_scores1(
    const float* __restrict__ xl, const float* __restrict__ xr,
    const int* __restrict__ ei, const float* __restrict__ att,
    float* __restrict__ ev, float* __restrict__ emax, int E, int ET) {
  const int lane = threadIdx.x & 31;
  const int e = blockIdx.x * 8 + (threadIdx.x >> 5);
  if (e >= ET) return;
  int s, d; edge_sd(e, E, ei, s, d);
  v4f vs = ((const v4f*)(xl + (size_t)s * FDIM))[lane];
  v4f vd = ((const v4f*)(xr + (size_t)d * FDIM))[lane];
  v4f va = ((const v4f*)att)[lane];
  float acc = 0.0f;
#pragma unroll
  for (int i = 0; i < 4; ++i) {
    float v = vs[i] + vd[i];
    v = v > 0.0f ? v : NEG_SLOPE * v;
    acc += v * va[i];
  }
#pragma unroll
  for (int off = 16; off > 0; off >>= 1) acc += __shfl_xor(acc, off, 32);
  if (lane == 0) { ev[e] = acc; atomicMaxF(&emax[d], acc); }
}

// generic: a = exp(e - emax[dst]); denom[dst] += a  (thread per edge)
__global__ void edge_alpha(float* __restrict__ ev, const float* __restrict__ emax,
                           float* __restrict__ den, const int* __restrict__ ei,
                           int E, int ET) {
  int e = blockIdx.x * blockDim.x + threadIdx.x;
  if (e >= ET) return;
  int d = (e < E) ? ei[E + e] : (e - E);
  float a = expf(ev[e] - emax[d]);
  ev[e] = a;
  atomicAdd(&den[d], a);
}

// layer-1 aggregate: wave per edge; lane loads b128, 4 atomic f32 adds
__global__ __launch_bounds__(256) void edge_aggr1(
    const float* __restrict__ xl, const float* __restrict__ ev,
    const float* __restrict__ den, const int* __restrict__ ei,
    float* __restrict__ gat, int E, int ET) {
  const int lane = threadIdx.x & 31;
  const int e = blockIdx.x * 8 + (threadIdx.x >> 5);
  if (e >= ET) return;
  int s, d; edge_sd(e, E, ei, s, d);
  const float w = ev[e] / den[d];
  v4f vs = ((const v4f*)(xl + (size_t)s * FDIM))[lane];
  float* pd = gat + (size_t)d * FDIM + lane * 4;
#pragma unroll
  for (int i = 0; i < 4; ++i)
    atomicAdd(&pd[i], w * vs[i]);
}

// h = relu(gat + b1 + skip + blin1), in place on gat; 4 elems/thread
__global__ void combine1(float* __restrict__ gat, const float* __restrict__ sk,
                         const float* __restrict__ b1, const float* __restrict__ blin1,
                         int total4) {
  int t = blockIdx.x * blockDim.x + threadIdx.x;
  if (t >= total4) return;
  int k4 = t & (FDIM / 4 - 1);
  v4f g  = ((v4f*)gat)[t];
  v4f s  = ((const v4f*)sk)[t];
  v4f bb = ((const v4f*)b1)[k4];
  v4f bl = ((const v4f*)blin1)[k4];
#pragma unroll
  for (int i = 0; i < 4; ++i) {
    float v = g[i] + bb[i] + s[i] + bl[i];
    g[i] = v > 0.0f ? v : 0.0f;
  }
  ((v4f*)gat)[t] = g;
}

// layer-2 node transforms (128 -> 2, three matrices): wave per node, b128 loads
__global__ __launch_bounds__(256) void node_xform2(
    const float* __restrict__ h, const float* __restrict__ Wl2,
    const float* __restrict__ Wr2, const float* __restrict__ Wlin2,
    float* __restrict__ xl2, float* __restrict__ xr2, float* __restrict__ sk2,
    int N) {
  const int lane = threadIdx.x & 31;
  const int n = blockIdx.x * 8 + (threadIdx.x >> 5);
  if (n >= N) return;
  v4f hv = ((const v4f*)(h + (size_t)n * FDIM))[lane];
  // rows k = lane*4 .. lane*4+3 of [128,2] weights -> 8 consecutive floats
  v4f la = ((const v4f*)Wl2)[lane * 2],   lb = ((const v4f*)Wl2)[lane * 2 + 1];
  v4f ra = ((const v4f*)Wr2)[lane * 2],   rb = ((const v4f*)Wr2)[lane * 2 + 1];
  v4f sa = ((const v4f*)Wlin2)[lane * 2], sb = ((const v4f*)Wlin2)[lane * 2 + 1];
  float a0 = hv[0]*la[0] + hv[1]*la[2] + hv[2]*lb[0] + hv[3]*lb[2];
  float a1 = hv[0]*la[1] + hv[1]*la[3] + hv[2]*lb[1] + hv[3]*lb[3];
  float r0 = hv[0]*ra[0] + hv[1]*ra[2] + hv[2]*rb[0] + hv[3]*rb[2];
  float r1 = hv[0]*ra[1] + hv[1]*ra[3] + hv[2]*rb[1] + hv[3]*rb[3];
  float s0 = hv[0]*sa[0] + hv[1]*sa[2] + hv[2]*sb[0] + hv[3]*sb[2];
  float s1 = hv[0]*sa[1] + hv[1]*sa[3] + hv[2]*sb[1] + hv[3]*sb[3];
#pragma unroll
  for (int off = 16; off > 0; off >>= 1) {
    a0 += __shfl_xor(a0, off, 32); a1 += __shfl_xor(a1, off, 32);
    r0 += __shfl_xor(r0, off, 32); r1 += __shfl_xor(r1, off, 32);
    s0 += __shfl_xor(s0, off, 32); s1 += __shfl_xor(s1, off, 32);
  }
  if (lane == 0) {
    xl2[n * 2] = a0; xl2[n * 2 + 1] = a1;
    xr2[n * 2] = r0; xr2[n * 2 + 1] = r1;
    sk2[n * 2] = s0; sk2[n * 2 + 1] = s1;
  }
}

// layer-2 edge score: thread per edge (C=2)
__global__ void edge_scores2(const float* __restrict__ xl2, const float* __restrict__ xr2,
                             const int* __restrict__ ei, const float* __restrict__ att2,
                             float* __restrict__ ev, float* __restrict__ emax2,
                             int E, int ET) {
  int e = blockIdx.x * blockDim.x + threadIdx.x;
  if (e >= ET) return;
  int s, d; edge_sd(e, E, ei, s, d);
  float m0 = xl2[s * 2]     + xr2[d * 2];
  float m1 = xl2[s * 2 + 1] + xr2[d * 2 + 1];
  m0 = m0 > 0.0f ? m0 : NEG_SLOPE * m0;
  m1 = m1 > 0.0f ? m1 : NEG_SLOPE * m1;
  float sc = m0 * att2[0] + m1 * att2[1];
  ev[e] = sc;
  atomicMaxF(&emax2[d], sc);
}

// layer-2 aggregate: thread per edge, 2 atomic adds
__global__ void edge_aggr2(const float* __restrict__ xl2, const float* __restrict__ ev,
                           const float* __restrict__ den2, const int* __restrict__ ei,
                           float* __restrict__ gat2, int E, int ET) {
  int e = blockIdx.x * blockDim.x + threadIdx.x;
  if (e >= ET) return;
  int s, d; edge_sd(e, E, ei, s, d);
  float w = ev[e] / den2[d];
  atomicAdd(&gat2[d * 2],     w * xl2[s * 2]);
  atomicAdd(&gat2[d * 2 + 1], w * xl2[s * 2 + 1]);
}

// out = log_softmax(gat2 + b2 + skip2 + blin2) over C=2
__global__ void finalize(const float* __restrict__ gat2, const float* __restrict__ sk2,
                         const float* __restrict__ b2, const float* __restrict__ blin2,
                         float* __restrict__ out, int N) {
  int n = blockIdx.x * blockDim.x + threadIdx.x;
  if (n >= N) return;
  float v0 = gat2[n * 2]     + b2[0] + sk2[n * 2]     + blin2[0];
  float v1 = gat2[n * 2 + 1] + b2[1] + sk2[n * 2 + 1] + blin2[1];
  float m = fmaxf(v0, v1);
  float lse = m + logf(expf(v0 - m) + expf(v1 - m));
  out[n * 2]     = v0 - lse;
  out[n * 2 + 1] = v1 - lse;
}

// ---------------------------------------------------------------------------

extern "C" void kernel_launch(void* const* d_in, const int* in_sizes, int n_in,
                              void* d_out, int out_size, void* d_ws, size_t ws_size,
                              hipStream_t stream) {
  const float* x     = (const float*)d_in[0];
  const int*   ei    = (const int*)d_in[1];
  const float* Wl1   = (const float*)d_in[2];
  const float* Wr1   = (const float*)d_in[3];
  const float* att1  = (const float*)d_in[4];
  const float* b1    = (const float*)d_in[5];
  const float* Wlin1 = (const float*)d_in[6];
  const float* blin1 = (const float*)d_in[7];
  const float* Wl2   = (const float*)d_in[8];
  const float* Wr2   = (const float*)d_in[9];
  const float* att2  = (const float*)d_in[10];
  const float* b2    = (const float*)d_in[11];
  const float* Wlin2 = (const float*)d_in[12];
  const float* blin2 = (const float*)d_in[13];

  const int N  = in_sizes[0] / FDIM;   // 50000
  const int E  = in_sizes[1] / 2;      // 800000
  const int ET = E + N;                // +self loops
  const size_t NF = (size_t)N * FDIM;

  // ---- workspace carve-up (floats) ----
  float* ws = (float*)d_ws;
  size_t o = 0;
  _Float16* xh = (_Float16*)(ws + o); o += NF / 2;            // f16 copy of x
  _Float16* Bp = (_Float16*)(ws + o); o += (3 * 8 * 4 * 32 * 16) / 2;
  float* xl1  = ws + o; o += NF;
  float* xr1  = ws + o; o += NF;
  float* sk1  = ws + o; o += NF;
  float* gat1 = ws + o; o += NF;      // becomes h in place
  float* ev   = ws + o; o += (size_t)ET;
  float* emax = ws + o; o += (size_t)N;
  float* den  = ws + o; o += (size_t)N;
  float* xl2  = ws + o; o += (size_t)N * 2;
  float* xr2  = ws + o; o += (size_t)N * 2;
  float* sk2  = ws + o; o += (size_t)N * 2;
  float* gat2 = ws + o; o += (size_t)N * 2;
  float* emax2 = ws + o; o += (size_t)N;
  float* den2  = ws + o; o += (size_t)N;

  const int TB = 256;
  const int nstrips = N / 16;

  // ---- init ----
  hipMemsetAsync(gat1, 0, NF * sizeof(float), stream);
  hipMemsetAsync(den,  0, (size_t)N * sizeof(float), stream);
  hipMemsetAsync(gat2, 0, (size_t)N * 2 * sizeof(float), stream);
  hipMemsetAsync(den2, 0, (size_t)N * sizeof(float), stream);
  fill_f32<<<(N + TB - 1) / TB, TB, 0, stream>>>(emax,  -INFINITY, N);
  fill_f32<<<(N + TB - 1) / TB, TB, 0, stream>>>(emax2, -INFINITY, N);

  // ---- f16 convert + weight pack ----
  const int NF4 = (int)(NF / 4);
  cvt_f16<<<(NF4 + TB - 1) / TB, TB, 0, stream>>>(x, xh, NF4);
  pack_b<<<(3 * 8 * 4 * 32 + TB - 1) / TB, TB, 0, stream>>>(Wl1, Wr1, Wlin1, Bp);

  // ---- layer-1 GEMMs (WMMA): xl1 = x@Wl1, xr1 = x@Wr1, sk1 = x@Wlin1 ----
  gemm3_wmma<<<(nstrips + 7) / 8, TB, 0, stream>>>(xh, Bp, xl1, xr1, sk1, nstrips);

  // ---- layer-1 attention ----
  const int ewaveB = (ET + 7) / 8;           // 8 waves/block, wave per edge
  const int ethrB  = (ET + TB - 1) / TB;     // thread per edge
  edge_scores1<<<ewaveB, TB, 0, stream>>>(xl1, xr1, ei, att1, ev, emax, E, ET);
  edge_alpha  <<<ethrB, TB, 0, stream>>>(ev, emax, den, ei, E, ET);
  edge_aggr1  <<<ewaveB, TB, 0, stream>>>(xl1, ev, den, ei, gat1, E, ET);
  combine1    <<<(NF4 + TB - 1) / TB, TB, 0, stream>>>(gat1, sk1, b1, blin1, NF4);

  // ---- layer-2 ----
  node_xform2 <<<(N + 7) / 8, TB, 0, stream>>>(gat1, Wl2, Wr2, Wlin2, xl2, xr2, sk2, N);
  edge_scores2<<<ethrB, TB, 0, stream>>>(xl2, xr2, ei, att2, ev, emax2, E, ET);
  edge_alpha  <<<ethrB, TB, 0, stream>>>(ev, emax2, den2, ei, E, ET);
  edge_aggr2  <<<ethrB, TB, 0, stream>>>(xl2, ev, den2, ei, gat2, E, ET);
  finalize    <<<(N + TB - 1) / TB, TB, 0, stream>>>(gat2, sk2, b2, blin2, (float*)d_out, N);

  // ---- second tuple element: edge_index passthrough into d_out tail ----
  long long tail_elems = (long long)out_size - (long long)N * 2;
  if (tail_elems > 0) {
    size_t bytes = (size_t)tail_elems * sizeof(float);
    size_t cap = (size_t)in_sizes[1] * 8;   // never read past int64 extent
    if (bytes > cap) bytes = cap;
    hipMemcpyAsync((float*)d_out + (size_t)N * 2, d_in[1], bytes,
                   hipMemcpyDeviceToDevice, stream);
  }
}
